// DHN_59811714564549
// MI455X (gfx1250) — matrix operand order repaired and checked by reading, not compiled, not run on hardware
//
#include <hip/hip_runtime.h>
#include <math.h>

#define EPSF 1e-15f
#define MAXNRM (1.0f - 1e-5f)
#define DIM 64

typedef __attribute__((ext_vector_type(16))) _Float16 v16h;
typedef __attribute__((ext_vector_type(8)))  float    v8f;

__device__ __forceinline__ float wsum(float v) {
    v += __shfl_xor(v, 1, 32);
    v += __shfl_xor(v, 2, 32);
    v += __shfl_xor(v, 4, 32);
    v += __shfl_xor(v, 8, 32);
    v += __shfl_xor(v, 16, 32);
    return v;
}

// ---------------- Stage 1: hyperbolic KG edge messages, one wave32 per edge ----
__global__ void kg_edge_kernel(const float* __restrict__ ent,
                               const float* __restrict__ rel,
                               const int* __restrict__ ksrc,
                               const int* __restrict__ kdst,
                               const int* __restrict__ ktyp,
                               float* __restrict__ accum,
                               float* __restrict__ cnt,
                               int E) {
    int wid  = blockIdx.x * (blockDim.x >> 5) + (threadIdx.x >> 5);
    int lane = threadIdx.x & 31;
    if (wid >= E) return;
    int is = ksrc[wid], id = kdst[wid], ir = ktyp[wid] + 2;

    float2 s = ((const float2*)(ent + (size_t)is * DIM))[lane];
    float2 d = ((const float2*)(ent + (size_t)id * DIM))[lane];
    float2 r = ((const float2*)(rel + (size_t)ir * DIM))[lane];

    // p = expmap0(s)
    float ss = wsum(s.x * s.x + s.y * s.y);
    float ns = sqrtf(fmaxf(ss, EPSF));
    float th = tanhf(ns);
    float fs = th / ns;
    float2 p = make_float2(s.x * fs, s.y * fs);
    float p2 = (th * th) * (ss / (ns * ns));
    float lam = 2.0f / fmaxf(1.0f - p2, EPSF);

    // dst_h = expmap(d, p)
    float2 X;
    {
        float uu = wsum(d.x * d.x + d.y * d.y);
        float pu = wsum(p.x * d.x + p.y * d.y);
        float nu = sqrtf(fmaxf(uu, EPSF));
        float c  = tanhf(0.5f * lam * nu) / nu;
        float w2 = c * c * uu, pw = c * pu;
        float a  = 1.0f + 2.0f * pw + w2;
        float b  = (1.0f - p2) * c;
        float inv = 1.0f / fmaxf(1.0f + 2.0f * pw + p2 * w2, EPSF);
        X = make_float2((a * p.x + b * d.x) * inv, (a * p.y + b * d.y) * inv);
    }
    // rel_h = expmap(r, p)
    float2 Y;
    {
        float uu = wsum(r.x * r.x + r.y * r.y);
        float pu = wsum(p.x * r.x + p.y * r.y);
        float nu = sqrtf(fmaxf(uu, EPSF));
        float c  = tanhf(0.5f * lam * nu) / nu;
        float w2 = c * c * uu, pw = c * pu;
        float a  = 1.0f + 2.0f * pw + w2;
        float b  = (1.0f - p2) * c;
        float inv = 1.0f / fmaxf(1.0f + 2.0f * pw + p2 * w2, EPSF);
        Y = make_float2((a * p.x + b * r.x) * inv, (a * p.y + b * r.y) * inv);
    }

    // m = mobius_add(dst_h, rel_h)
    float x2 = wsum(X.x * X.x + X.y * X.y);
    float y2 = wsum(Y.x * Y.x + Y.y * Y.y);
    float xy = wsum(X.x * Y.x + X.y * Y.y);
    float a1 = 1.0f + 2.0f * xy + y2;
    float b1 = 1.0f - x2;
    float inv1 = 1.0f / fmaxf(1.0f + 2.0f * xy + x2 * y2, EPSF);
    float2 m = make_float2((a1 * X.x + b1 * Y.x) * inv1, (a1 * X.y + b1 * Y.y) * inv1);

    // project
    float mm = wsum(m.x * m.x + m.y * m.y);
    float nm = sqrtf(fmaxf(mm, EPSF));
    float f  = (nm > MAXNRM) ? (MAXNRM / nm) : 1.0f;
    m.x *= f; m.y *= f;
    float m2 = mm * f * f;

    // logmap(m, p): sub = mobius_add(-p, m)
    float pm  = wsum(p.x * m.x + p.y * m.y);
    float nxy = -pm;
    float a2  = 1.0f + 2.0f * nxy + m2;   // coeff on (-p)
    float b2  = 1.0f - p2;                // coeff on m
    float inv2 = 1.0f / fmaxf(1.0f + 2.0f * nxy + p2 * m2, EPSF);
    float2 sub = make_float2((-a2 * p.x + b2 * m.x) * inv2,
                             (-a2 * p.y + b2 * m.y) * inv2);
    float s2 = wsum(sub.x * sub.x + sub.y * sub.y);
    float nsub = sqrtf(fmaxf(s2, EPSF));
    float scale = (2.0f / lam) * atanhf(fminf(nsub, MAXNRM)) / nsub;

    float* dst = accum + (size_t)is * DIM + 2 * lane;
    atomicAdd(dst,     sub.x * scale);
    atomicAdd(dst + 1, sub.y * scale);
    if (lane == 0) atomicAdd(&cnt[is], 1.0f);
}

// ---------------- Stage 2: finalize seg_mean, scatter to outputs -------------
__global__ void finalize_out_kernel(float* __restrict__ accum,
                                    const float* __restrict__ cnt,
                                    float* __restrict__ out1,
                                    float* __restrict__ out3,
                                    int n_ent, int n_items) {
    int i = blockIdx.x * blockDim.x + threadIdx.x;
    if (i >= n_ent * DIM) return;
    int row = i >> 6;
    float v = accum[i] / fmaxf(cnt[row], 1.0f);
    accum[i] = v;                               // accum becomes `out`
    if (row >= n_items) out1[i] = v;            // out[n_items:] part of output 1
    else out3[(size_t)row * DIM + (i & 63)] = v; // out_items (output 3)
}

// ---------------- Stage 3: row squared norms of cat_nodes --------------------
__global__ void norm_kernel(const float* __restrict__ out,
                            const float* __restrict__ usr,
                            float* __restrict__ nall,
                            int n_items, int N) {
    int row  = blockIdx.x * (blockDim.x >> 5) + (threadIdx.x >> 5);
    int lane = threadIdx.x & 31;
    if (row >= N) return;
    const float* base = (row < n_items) ? (out + (size_t)row * DIM)
                                        : (usr + (size_t)(row - n_items) * DIM);
    float2 v = ((const float2*)base)[lane];
    float s = wsum(v.x * v.x + v.y * v.y);
    if (lane == 0) nall[row] = s;
}

// ---------------- Stage 4: item->user aggregation ----------------------------
__global__ void iu_kernel(const float* __restrict__ out,
                          const float* __restrict__ usr,
                          const int* __restrict__ esrc,
                          const int* __restrict__ edst,
                          float* __restrict__ uacc,
                          float* __restrict__ ucnt,
                          int n_items, int n_usr, int E) {
    int wid  = blockIdx.x * (blockDim.x >> 5) + (threadIdx.x >> 5);
    int lane = threadIdx.x & 31;
    if (wid >= E) return;
    int srow = esrc[wid];
    int drow = edst[wid] - n_items;
    if (drow < 0 || drow >= n_usr) return;
    const float* base = (srow < n_items) ? (out + (size_t)srow * DIM)
                                         : (usr + (size_t)(srow - n_items) * DIM);
    float2 v = ((const float2*)base)[lane];
    float* dst = uacc + (size_t)drow * DIM + 2 * lane;
    atomicAdd(dst,     v.x);
    atomicAdd(dst + 1, v.y);
    if (lane == 0) atomicAdd(&ucnt[drow], 1.0f);
}

__global__ void u_finalize_kernel(const float* __restrict__ uacc,
                                  const float* __restrict__ ucnt,
                                  float* __restrict__ out2, int n_usr) {
    int i = blockIdx.x * blockDim.x + threadIdx.x;
    if (i >= n_usr * DIM) return;
    out2[i] = uacc[i] / fmaxf(ucnt[i >> 6], 1.0f);
}

// ---------------- Stage 5: user->item normalized aggregation -----------------
__global__ void ui_kernel(const float* __restrict__ out,
                          const float* __restrict__ usr,
                          const float* __restrict__ nall,
                          const int* __restrict__ esrc,
                          const int* __restrict__ edst,
                          float* __restrict__ icf,
                          float* __restrict__ icnt,
                          int n_items, int E) {
    int wid  = blockIdx.x * (blockDim.x >> 5) + (threadIdx.x >> 5);
    int lane = threadIdx.x & 31;
    if (wid >= E) return;
    int srow = esrc[wid];
    int drow = edst[wid];
    if (drow < 0 || drow >= n_items) return;
    const float* base = (srow < n_items) ? (out + (size_t)srow * DIM)
                                         : (usr + (size_t)(srow - n_items) * DIM);
    float2 v = ((const float2*)base)[lane];
    float sc = nall[drow] / (nall[srow] + 1e-6f);
    float* dst = icf + (size_t)drow * DIM + 2 * lane;
    atomicAdd(dst,     v.x * sc);
    atomicAdd(dst + 1, v.y * sc);
    if (lane == 0) atomicAdd(&icnt[drow], 1.0f);
}

__global__ void icf_finalize_kernel(float* __restrict__ icf,
                                    const float* __restrict__ icnt, int n_items) {
    int i = blockIdx.x * blockDim.x + threadIdx.x;
    if (i >= n_items * DIM) return;
    icf[i] = icf[i] / fmaxf(icnt[i >> 6], 1.0f);
}

// ---------------- Stage 6: WMMA gated fusion ---------------------------------
__device__ __forceinline__ v8f wmma_f16(v16h a, v16h b, v8f c) {
    return __builtin_amdgcn_wmma_f32_16x16x32_f16(
        /*neg_a=*/false, a, /*neg_b=*/false, b,
        /*c_mod=*/(short)0, c, /*reuse_a=*/false, /*reuse_b=*/false);
}

// gi = sigmoid(out_items @ W1.T + i_cf @ W2.T); fusion = gi*oi + (1-gi)*icf
// one wave32 per 16-row tile; K=64 split in two 16x16x32 WMMAs per operand.
__global__ void gemm_gate_kernel(const float* __restrict__ oitems,
                                 const float* __restrict__ icf,
                                 const float* __restrict__ W1,
                                 const float* __restrict__ W2,
                                 float* __restrict__ fusion,
                                 int n_items) {
    int m0   = blockIdx.x * 16;
    int lane = threadIdx.x & 31;
    int lo = lane & 15, hi = lane >> 4;

    int mr = m0 + lo;
    if (mr > n_items - 1) mr = n_items - 1;     // clamp (keep EXEC all-ones)
    const float* rowE = oitems + (size_t)mr * DIM;
    const float* rowC = icf    + (size_t)mr * DIM;

    // A fragment layout (16-bit A 16x32): VGPR v, lane group hi:
    //   K = (v/4)*16 + hi*8 + (v%4)*2 + j
    v16h aE0, aE1, aC0, aC1;
#pragma unroll
    for (int v = 0; v < 8; ++v) {
        int kb = (v >> 2) * 16 + hi * 8 + (v & 3) * 2;
#pragma unroll
        for (int j = 0; j < 2; ++j) {
            aE0[2 * v + j] = (_Float16)rowE[kb + j];
            aE1[2 * v + j] = (_Float16)rowE[32 + kb + j];
            aC0[2 * v + j] = (_Float16)rowC[kb + j];
            aC1[2 * v + j] = (_Float16)rowC[32 + kb + j];
        }
    }

#pragma unroll
    for (int nt = 0; nt < 4; ++nt) {
        int n = nt * 16 + lo;                   // output column for this lane
        // B fragment layout (16-bit B 32x16): same K order as A,
        //   row k = (v/4)*16 + hi*8 + (v%4)*2 + j, column n = lane&15.  B[k][n] = W[n][k].
        v16h b10, b11, b20, b21;
#pragma unroll
        for (int v = 0; v < 8; ++v) {
#pragma unroll
            for (int j = 0; j < 2; ++j) {
                int k = (v >> 2) * 16 + hi * 8 + (v & 3) * 2 + j;
                b10[2 * v + j] = (_Float16)W1[(size_t)n * DIM + k];
                b11[2 * v + j] = (_Float16)W1[(size_t)n * DIM + 32 + k];
                b20[2 * v + j] = (_Float16)W2[(size_t)n * DIM + k];
                b21[2 * v + j] = (_Float16)W2[(size_t)n * DIM + 32 + k];
            }
        }
        v8f c = {};
        c = wmma_f16(aE0, b10, c);
        c = wmma_f16(aE1, b11, c);
        c = wmma_f16(aC0, b20, c);
        c = wmma_f16(aC1, b21, c);
        // C/D layout: VGPR r -> M = m0 + 8*hi + r, N = nt*16 + (lane&15)
#pragma unroll
        for (int rr = 0; rr < 8; ++rr) {
            int m = m0 + hi * 8 + rr;
            if (m < n_items) {
                int col = nt * 16 + lo;
                float oi = oitems[(size_t)m * DIM + col];
                float ic = icf[(size_t)m * DIM + col];
                float g  = 1.0f / (1.0f + expf(-c[rr]));
                fusion[(size_t)m * DIM + col] = g * oi + (1.0f - g) * ic;
            }
        }
    }
}

// ---------------- Launcher ---------------------------------------------------
extern "C" void kernel_launch(void* const* d_in, const int* in_sizes, int n_in,
                              void* d_out, int out_size, void* d_ws, size_t ws_size,
                              hipStream_t stream) {
    const float* ent = (const float*)d_in[0];
    const float* usr = (const float*)d_in[1];
    const float* rel = (const float*)d_in[2];
    const float* W1  = (const float*)d_in[3];
    const float* W2  = (const float*)d_in[4];
    const int* ksrc  = (const int*)d_in[5];
    const int* kdst  = (const int*)d_in[6];
    const int* ktyp  = (const int*)d_in[7];
    const int* iusrc = (const int*)d_in[8];
    const int* iudst = (const int*)d_in[9];
    const int* uisrc = (const int*)d_in[10];
    const int* uidst = (const int*)d_in[11];

    const int n_ent   = in_sizes[0] / DIM;
    const int n_usr   = in_sizes[1] / DIM;
    const int E_KG    = in_sizes[5];
    const int E_IU    = in_sizes[8];
    const int E_UI    = in_sizes[10];
    const int n_items = out_size / DIM - n_ent - n_usr;   // derived, no D2H copy
    const int N       = n_items + n_usr;

    float* out1 = (float*)d_out;                 // [n_ent, 64]: fusion ++ out[n_items:]
    float* out2 = out1 + (size_t)n_ent * DIM;    // [n_usr, 64]: u
    float* out3 = out2 + (size_t)n_usr * DIM;    // [n_items, 64]: out_items

    float* ws = (float*)d_ws;
    size_t o = 0;
    float* accum = ws + o; o += (size_t)n_ent * DIM;
    float* cnt   = ws + o; o += (size_t)n_ent;
    float* uacc  = ws + o; o += (size_t)n_usr * DIM;
    float* ucnt  = ws + o; o += (size_t)n_usr;
    float* icf   = ws + o; o += (size_t)n_items * DIM;
    float* icnt  = ws + o; o += (size_t)n_items;
    float* nall  = ws + o; o += (size_t)N;

    size_t zbytes = o * sizeof(float);
    if (zbytes > ws_size) zbytes = ws_size;
    hipMemsetAsync(d_ws, 0, zbytes, stream);     // graph-capture-legal memset node

    const int WPB = 8;                           // 8 wave32 per 256-thread block
    kg_edge_kernel<<<(E_KG + WPB - 1) / WPB, 256, 0, stream>>>(
        ent, rel, ksrc, kdst, ktyp, accum, cnt, E_KG);

    int totE = n_ent * DIM;
    finalize_out_kernel<<<(totE + 255) / 256, 256, 0, stream>>>(
        accum, cnt, out1, out3, n_ent, n_items);

    norm_kernel<<<(N + WPB - 1) / WPB, 256, 0, stream>>>(
        accum, usr, nall, n_items, N);

    iu_kernel<<<(E_IU + WPB - 1) / WPB, 256, 0, stream>>>(
        accum, usr, iusrc, iudst, uacc, ucnt, n_items, n_usr, E_IU);

    int totU = n_usr * DIM;
    u_finalize_kernel<<<(totU + 255) / 256, 256, 0, stream>>>(
        uacc, ucnt, out2, n_usr);

    ui_kernel<<<(E_UI + WPB - 1) / WPB, 256, 0, stream>>>(
        accum, usr, nall, uisrc, uidst, icf, icnt, n_items, E_UI);

    int totI = n_items * DIM;
    icf_finalize_kernel<<<(totI + 255) / 256, 256, 0, stream>>>(icf, icnt, n_items);

    gemm_gate_kernel<<<(n_items + 15) / 16, 32, 0, stream>>>(
        accum, icf, W1, W2, out1, n_items);

    (void)n_in; (void)ws_size;
}
